// KHopGraphConvolution_38826504356275
// MI455X (gfx1250) — matrix operand order repaired and checked by reading, not compiled, not run on hardware
//
#include <hip/hip_runtime.h>

#define N_NODES 16384
#define FDIM    64
#define KSLICES 4

typedef __attribute__((ext_vector_type(16))) __bf16        v16bf;
typedef __attribute__((ext_vector_type(8)))  float         v8f;
typedef __attribute__((ext_vector_type(4)))  float         v4f;
typedef __attribute__((ext_vector_type(4)))  unsigned int  v4u;

union BF16Vec {
    v16bf bf;
    v4u   u[2];
};

static __device__ __forceinline__ unsigned short f32_to_bf16_bits(float f) {
    union { __bf16 h; unsigned short s; } u;
    u.h = (__bf16)f;
    return u.s;
}

// B tile slice (32x16 bf16): lane = column n (lane&15); lanes 0-15 carry
// K=0..15, lanes 16-31 carry K=16..31 -> one contiguous 32 B chunk per lane.
static __device__ __forceinline__ v16bf load_b_tile(const unsigned short* p) {
    BF16Vec b;
    b.u[0] = *(const v4u*)(p);
    b.u[1] = *(const v4u*)(p + 8);
    return b.bf;
}

// Pack 16 streamed f32 of L into the bf16 A fragment (16x32 layout:
// lane = row M, half-wave selects K 0-15 vs 16-31).
static __device__ __forceinline__ v16bf cvt_a(v4f a0, v4f a1, v4f a2, v4f a3) {
    v16bf a;
#pragma unroll
    for (int i = 0; i < 4; ++i) {
        a[i]      = (__bf16)a0[i];
        a[4 + i]  = (__bf16)a1[i];
        a[8 + i]  = (__bf16)a2[i];
        a[12 + i] = (__bf16)a3[i];
    }
    return a;
}

// ---------------------------------------------------------------------------
// Kernel 1: x (N x 64 f32) -> Tt (64 x N bf16) transposed copy
// ---------------------------------------------------------------------------
__global__ __launch_bounds__(256)
void cheb_cvt_transpose(const float* __restrict__ x, unsigned short* __restrict__ Tt) {
    int idx = blockIdx.x * 256 + threadIdx.x;
    int n = idx >> 6;
    int f = idx & 63;
    Tt[(size_t)f * N_NODES + n] = f32_to_bf16_bits(x[idx]);
}

// ---------------------------------------------------------------------------
// Kernel 2: partial GEMM  Ypart[slice] = L[:, ks:ke] @ T_in[ks:ke, :]
// Each wave: one 16-row tile x all 64 cols (4 C tiles); 8 waves/block;
// grid = (N/128 row blocks, KSLICES k-slices). Software-pipelined: next
// chunk's A/B loads are in flight across the current chunk's 4 WMMAs.
// ---------------------------------------------------------------------------
__global__ __launch_bounds__(256)
void cheb_gemm_partial(const float* __restrict__ L,
                       const unsigned short* __restrict__ TtIn,
                       float* __restrict__ Ypart) {
    const int tid  = threadIdx.x;
    const int wave = tid >> 5;
    const int lane = tid & 31;
    const int lo   = lane & 15;
    const int hi   = lane >> 4;
    const int rowbase = blockIdx.x * 128 + wave * 16;

    const int KS = N_NODES / KSLICES;
    const int ks = blockIdx.y * KS;
    const int ke = ks + KS;

    const float* Lrow = L + (size_t)(rowbase + lo) * N_NODES;
    const unsigned short* Bt0 = TtIn + (size_t)(0  + lo) * N_NODES;
    const unsigned short* Bt1 = TtIn + (size_t)(16 + lo) * N_NODES;
    const unsigned short* Bt2 = TtIn + (size_t)(32 + lo) * N_NODES;
    const unsigned short* Bt3 = TtIn + (size_t)(48 + lo) * N_NODES;

    v8f acc0 = {}, acc1 = {}, acc2 = {}, acc3 = {};

    // ---- prologue: load chunk ks ----
    int aoff = ks + hi * 16;
    v4f p0 = *(const v4f*)(Lrow + aoff);
    v4f p1 = *(const v4f*)(Lrow + aoff + 4);
    v4f p2 = *(const v4f*)(Lrow + aoff + 8);
    v4f p3 = *(const v4f*)(Lrow + aoff + 12);
    int boff = ks + hi * 16;
    v16bf bc0 = load_b_tile(Bt0 + boff);
    v16bf bc1 = load_b_tile(Bt1 + boff);
    v16bf bc2 = load_b_tile(Bt2 + boff);
    v16bf bc3 = load_b_tile(Bt3 + boff);
    v16bf ac  = cvt_a(p0, p1, p2, p3);

#pragma unroll 2
    for (int kc = ks + 32; kc < ke; kc += 32) {
        // ---- issue next chunk's loads (A stream from HBM, B from L2) ----
        const int an = kc + hi * 16;
        v4f n0 = *(const v4f*)(Lrow + an);
        v4f n1 = *(const v4f*)(Lrow + an + 4);
        v4f n2 = *(const v4f*)(Lrow + an + 8);
        v4f n3 = *(const v4f*)(Lrow + an + 12);
        __builtin_prefetch(Lrow + an + 512, 0, 1);
        const int bn = kc + hi * 16;
        v16bf bn0 = load_b_tile(Bt0 + bn);
        v16bf bn1 = load_b_tile(Bt1 + bn);
        v16bf bn2 = load_b_tile(Bt2 + bn);
        v16bf bn3 = load_b_tile(Bt3 + bn);

        // ---- compute current chunk ----
        acc0 = __builtin_amdgcn_wmma_f32_16x16x32_bf16(false, ac, false, bc0,
                                                       (short)0, acc0, false, false);
        acc1 = __builtin_amdgcn_wmma_f32_16x16x32_bf16(false, ac, false, bc1,
                                                       (short)0, acc1, false, false);
        acc2 = __builtin_amdgcn_wmma_f32_16x16x32_bf16(false, ac, false, bc2,
                                                       (short)0, acc2, false, false);
        acc3 = __builtin_amdgcn_wmma_f32_16x16x32_bf16(false, ac, false, bc3,
                                                       (short)0, acc3, false, false);

        // ---- rotate (renamed away by the unroll) ----
        ac  = cvt_a(n0, n1, n2, n3);
        bc0 = bn0; bc1 = bn1; bc2 = bn2; bc3 = bn3;
    }

    // ---- tail chunk ----
    acc0 = __builtin_amdgcn_wmma_f32_16x16x32_bf16(false, ac, false, bc0,
                                                   (short)0, acc0, false, false);
    acc1 = __builtin_amdgcn_wmma_f32_16x16x32_bf16(false, ac, false, bc1,
                                                   (short)0, acc1, false, false);
    acc2 = __builtin_amdgcn_wmma_f32_16x16x32_bf16(false, ac, false, bc2,
                                                   (short)0, acc2, false, false);
    acc3 = __builtin_amdgcn_wmma_f32_16x16x32_bf16(false, ac, false, bc3,
                                                   (short)0, acc3, false, false);

    // ---- store partial Y (C layout: VGPR r -> rows r / r+8, col = tile*16+lo)
    float* Yp = Ypart + (size_t)blockIdx.y * N_NODES * FDIM;
#pragma unroll
    for (int r = 0; r < 8; ++r) {
        const size_t base = (size_t)(rowbase + r + hi * 8) * FDIM + lo;
        Yp[base]      = acc0[r];
        Yp[base + 16] = acc1[r];
        Yp[base + 32] = acc2[r];
        Yp[base + 48] = acc3[r];
    }
}

// ---------------------------------------------------------------------------
// Kernel 3: combine K-slices + Chebyshev recurrence.
//   Y = sum_s Ypart[s];  T = firstPass ? Y : 2*Y - Tprev
// Writes T f32 (N x 64) and optionally bf16 transposed (64 x N).
// ---------------------------------------------------------------------------
__global__ __launch_bounds__(256)
void cheb_combine(const float* __restrict__ Ypart,
                  const float* __restrict__ Tprev,
                  float* __restrict__ TOut,
                  unsigned short* __restrict__ TtOut,
                  int firstPass, int writeTt) {
    const int idx = blockIdx.x * 256 + threadIdx.x;   // 0 .. N*64-1
    float y = Ypart[idx];
#pragma unroll
    for (int s = 1; s < KSLICES; ++s)
        y += Ypart[(size_t)s * N_NODES * FDIM + idx];

    float t = firstPass ? y : (2.0f * y - Tprev[idx]);
    TOut[idx] = t;
    if (writeTt) {
        const int n = idx >> 6;
        const int f = idx & 63;
        TtOut[(size_t)f * N_NODES + n] = f32_to_bf16_bits(t);
    }
}

// ---------------------------------------------------------------------------
// Kernel 4: out = sum_k T_k @ W_k + bias   (f32; ~0.5 GFLOP, L2-resident)
// ---------------------------------------------------------------------------
__global__ __launch_bounds__(256)
void cheb_project(const float* __restrict__ T0, const float* __restrict__ T1,
                  const float* __restrict__ T2, const float* __restrict__ T3,
                  const float* __restrict__ W,  const float* __restrict__ bias,
                  float* __restrict__ out) {
    const int tid = threadIdx.x;
    const int n   = blockIdx.x * 64 + (tid >> 2);
    const int q   = (tid & 3) * 16;

    float acc[16];
#pragma unroll
    for (int j = 0; j < 16; ++j) acc[j] = bias[q + j];

    const float* Ts[4] = {T0, T1, T2, T3};
#pragma unroll
    for (int k = 0; k < 4; ++k) {
        const float* Tr = Ts[k] + (size_t)n * FDIM;
        const float* Wk = W + k * FDIM * FDIM + q;
        for (int c = 0; c < FDIM; ++c) {
            const float t = Tr[c];
            const v4f* w = (const v4f*)(Wk + c * FDIM);
            v4f w0 = w[0], w1 = w[1], w2 = w[2], w3 = w[3];
#pragma unroll
            for (int j = 0; j < 4; ++j) {
                acc[j]      += t * w0[j];
                acc[4 + j]  += t * w1[j];
                acc[8 + j]  += t * w2[j];
                acc[12 + j] += t * w3[j];
            }
        }
    }
#pragma unroll
    for (int j = 0; j < 16; ++j) out[(size_t)n * FDIM + q + j] = acc[j];
}

// ---------------------------------------------------------------------------
extern "C" void kernel_launch(void* const* d_in, const int* in_sizes, int n_in,
                              void* d_out, int out_size, void* d_ws, size_t ws_size,
                              hipStream_t stream) {
    (void)in_sizes; (void)n_in; (void)out_size; (void)ws_size;

    const float* x    = (const float*)d_in[0];   // (N, 64)
    const float* L    = (const float*)d_in[1];   // (N, N)
    const float* W    = (const float*)d_in[2];   // (4, 64, 64)
    const float* bias = (const float*)d_in[3];   // (64,)
    float*       out  = (float*)d_out;           // (N, 64)

    // Workspace (~32 MB):
    //   TtA, TtB : bf16 64 x N        (2 MB each)   transposed-T ping-pong
    //   T1,T2,T3 : f32  N x 64        (4 MB each)   f32 T for projection
    //   Ypart    : f32  KSLICES x N x 64 (16 MB)    per-slice partials
    const size_t TT_ELEMS = (size_t)FDIM * N_NODES;
    const size_t T_ELEMS  = (size_t)N_NODES * FDIM;
    char* ws = (char*)d_ws;
    unsigned short* TtA = (unsigned short*)ws;
    unsigned short* TtB = (unsigned short*)(ws + TT_ELEMS * sizeof(unsigned short));
    float* T1    = (float*)(ws + 2 * TT_ELEMS * sizeof(unsigned short));
    float* T2    = T1 + T_ELEMS;
    float* T3    = T2 + T_ELEMS;
    float* Ypart = T3 + T_ELEMS;

    const dim3 ggrid(N_NODES / 128, KSLICES);
    const dim3 cgrid((N_NODES * FDIM) / 256);

    // T0 = x (bf16, transposed) for the first hop's B operand.
    cheb_cvt_transpose<<<cgrid, 256, 0, stream>>>(x, TtA);

    // T1 = L @ x
    cheb_gemm_partial<<<ggrid, 256, 0, stream>>>(L, TtA, Ypart);
    cheb_combine<<<cgrid, 256, 0, stream>>>(Ypart, x, T1, TtB, 1, 1);

    // T2 = 2*(L @ T1) - x
    cheb_gemm_partial<<<ggrid, 256, 0, stream>>>(L, TtB, Ypart);
    cheb_combine<<<cgrid, 256, 0, stream>>>(Ypart, x, T2, TtA, 0, 1);

    // T3 = 2*(L @ T2) - T1  (last hop: skip transposed store)
    cheb_gemm_partial<<<ggrid, 256, 0, stream>>>(L, TtA, Ypart);
    cheb_combine<<<cgrid, 256, 0, stream>>>(Ypart, T1, T3, TtB, 0, 0);

    // out = x@W0 + T1@W1 + T2@W2 + T3@W3 + bias
    cheb_project<<<dim3(N_NODES / 64), 256, 0, stream>>>(x, T1, T2, T3, W, bias, out);
}
